// GAT_TS1_83631603187930
// MI455X (gfx1250) — matrix operand order repaired and checked by reading, not compiled, not run on hardware
//
#include <hip/hip_runtime.h>
#include <hip/hip_bf16.h>

typedef _Float16 f16;
typedef __attribute__((ext_vector_type(16))) _Float16 v16h;
typedef __attribute__((ext_vector_type(8)))  _Float16 v8h;
typedef __attribute__((ext_vector_type(8)))  float    v8f;

// D = A(16x32 f16) * B(32x16 f16) + C(16x16 f32)
__device__ __forceinline__ v8f wmma_f16(v16h a, v16h b, v8f c) {
    return __builtin_amdgcn_wmma_f32_16x16x32_f16(false, a, false, b, (short)0, c, false, false);
}

__device__ __forceinline__ v8f bcast8(float v) {
    v8f r;
#pragma unroll
    for (int i = 0; i < 8; ++i) r[i] = v;
    return r;
}

// fast transcendentals: v_exp_f32 + raw v_rcp_f32 (TRANS ops co-execute with WMMA)
__device__ __forceinline__ float sigm(float x) {
    return __builtin_amdgcn_rcpf(1.0f + __expf(-x));
}
__device__ __forceinline__ float tanh_fast(float x) {
    return 1.0f - 2.0f * __builtin_amdgcn_rcpf(__expf(2.0f * x) + 1.0f);  // safe at +-inf
}
__device__ __forceinline__ float lrelu(float x) { return x > 0.0f ? x : 0.01f * x; }

// ---------------- constants ----------------
static constexpr int CB = 4;      // batch
static constexpr int CT = 60;     // time
static constexpr int CN = 2048;   // nodes
static constexpr int CH = 64;     // hidden
static constexpr int CG = 192;    // 3*H gates

// =====================================================================
// prep: f32 -> f16 weight conversion (B-operand layout = plain row major),
// plus folded GAT score vectors va1/va2 and scalars c1/c2.
// =====================================================================
__global__ __launch_bounds__(256)
void prep_kernel(const float* __restrict__ Wih0, const float* __restrict__ Whh0,
                 const float* __restrict__ Wih1, const float* __restrict__ Whh1,
                 const float* __restrict__ transW, const float* __restrict__ transb,
                 const float* __restrict__ a,
                 f16* __restrict__ Wih0h, f16* __restrict__ Whh0h,
                 f16* __restrict__ Wih1h, f16* __restrict__ Whh1h,
                 float* __restrict__ va1, float* __restrict__ va2, float* __restrict__ c12)
{
    int tid = threadIdx.x;
    for (int i = tid; i < CG * CH; i += 256) {
        Whh0h[i] = (f16)Whh0[i];
        Wih1h[i] = (f16)Wih1[i];
        Whh1h[i] = (f16)Whh1[i];
    }
    for (int i = tid; i < CG * 32; i += 256) {        // K padded 6 -> 32
        int g = i >> 5, k = i & 31;
        Wih0h[i] = (k < 6) ? (f16)Wih0[g * 6 + k] : (f16)0.0f;
    }
    if (tid < 64) {
        float s1 = 0.0f, s2 = 0.0f;
        for (int g = 0; g < 64; ++g) {
            s1 += a[g]      * transW[g * 64 + tid];
            s2 += a[64 + g] * transW[g * 64 + tid];
        }
        va1[tid] = s1; va2[tid] = s2;
    }
    if (tid == 0) {
        float s1 = 0.0f, s2 = 0.0f;
        for (int g = 0; g < 64; ++g) { s1 += transb[g] * a[g]; s2 += transb[g] * a[64 + g]; }
        c12[0] = s1; c12[1] = s2;
    }
}

// =====================================================================
// Fused 2-layer GRU over T=60 steps, 16 sequences / workgroup, 4 waves.
// Wave w owns gate columns {w*16 .. w*16+15} of r, z, n  -> wave-local
// nonlinearity + state update. State kept twice in LDS: f32 master (for
// the z*h_old blend + fc_in epilogue) and f16 shadow in WMMA-A-ready
// halves so each A build is two 16B ds_loads with zero conversions.
// fc_in fused into epilogue.
// =====================================================================
__global__ __launch_bounds__(128)
void gru_fused_kernel(const float* __restrict__ x,
                      const f16* __restrict__ Wih0h, const f16* __restrict__ Whh0h,
                      const f16* __restrict__ Wih1h, const f16* __restrict__ Whh1h,
                      const float* __restrict__ bih0, const float* __restrict__ bhh0,
                      const float* __restrict__ bih1, const float* __restrict__ bhh1,
                      const float* __restrict__ fcW,  const float* __restrict__ fcb,
                      float* __restrict__ hidden_out)
{
    __shared__ float h0s[16][68];   // f32 master state (pad 68: conflict-free columns)
    __shared__ float h1s[16][68];
    __shared__ f16   h0h[16][72];   // f16 shadow (pad 72 halves = 36 dwords)
    __shared__ f16   h1h[16][72];
    __shared__ float xbuf[16][8];

    const int tid  = threadIdx.x;
    const int lane = tid & 31;
    const int wv   = tid >> 5;
    const int s0   = blockIdx.x * 16;
    const int bb   = s0 / CN;
    const int n0   = s0 % CN;

    for (int i = tid; i < 16 * 68; i += 128) { (&h0s[0][0])[i] = 0.0f; (&h1s[0][0])[i] = 0.0f; }
    for (int i = tid; i < 16 * 72; i += 128) { (&h0h[0][0])[i] = (f16)0.0f; (&h1h[0][0])[i] = (f16)0.0f; }
    __syncthreads();

    const int mA  = lane & 15;          // A row  (sequence)
    const int kbA = (lane >> 4) * 16;   // A K sub-base
    const int nB  = lane & 15;          // B / C column
    const int kbB = (lane >> 4) * 16;   // B K sub-base

    const int gr = wv * 16 + nB;        // r col
    const int gz = 64 + gr;             // z col
    const int gn = 128 + gr;            // n col
    const float b0r = bih0[gr] + bhh0[gr], b0z = bih0[gz] + bhh0[gz];
    const float b0xn = bih0[gn], b0hn = bhh0[gn];
    const float b1r = bih1[gr] + bhh1[gr], b1z = bih1[gz] + bhh1[gz];
    const float b1xn = bih1[gn], b1hn = bhh1[gn];

    // A operand straight from the f16 shadow: two aligned 16B LDS loads.
    auto buildA = [&](const f16 (*hf)[72], int c) -> v16h {
        v8h lo = *(const v8h*)&hf[mA][c * 32 + kbA];
        v8h hi = *(const v8h*)&hf[mA][c * 32 + kbA + 8];
        v16h r;
#pragma unroll
        for (int i = 0; i < 8; ++i) { r[i] = lo[i]; r[8 + i] = hi[i]; }
        return r;
    };
    auto buildAx = [&]() -> v16h {      // x tile: K=6 (rest zero)
        v16h r;
#pragma unroll
        for (int i = 0; i < 8; ++i) { int k = kbA + i; r[i] = (k < 6) ? (f16)xbuf[mA][k] : (f16)0.0f; }
#pragma unroll
        for (int i = 0; i < 8; ++i) r[8 + i] = (f16)0.0f;
        return r;
    };
    auto loadB64 = [&](const f16* W, int g0, int c) -> v16h {   // W row-major [192][64]
        return *(const v16h*)(W + (size_t)(g0 + nB) * 64 + c * 32 + kbB);
    };
    auto loadB32 = [&](const f16* W, int g0) -> v16h {          // W row-major [192][32]
        return *(const v16h*)(W + (size_t)(g0 + nB) * 32 + kbB);
    };

    const int g0r = wv * 16, g0z = 64 + wv * 16, g0n = 128 + wv * 16;

    for (int t = 0; t < CT; ++t) {
        // stage x_t: 16 consecutive nodes x 6 features = 96 contiguous floats
        const float* xp = x + (((size_t)bb * CT + t) * CN + n0) * 6;
        if (tid < 96) xbuf[tid / 6][tid % 6] = xp[tid];
        __syncthreads();

        // ---------------- layer 0 ----------------
        v8f ar = bcast8(b0r), az = bcast8(b0z), axn = bcast8(b0xn), ahn = bcast8(b0hn);
        {
            v16h axv = buildAx();
            ar  = wmma_f16(axv, loadB32(Wih0h, g0r), ar);
            az  = wmma_f16(axv, loadB32(Wih0h, g0z), az);
            axn = wmma_f16(axv, loadB32(Wih0h, g0n), axn);
            v16h a0 = buildA(h0h, 0), a1 = buildA(h0h, 1);
            ar  = wmma_f16(a0, loadB64(Whh0h, g0r, 0), ar);
            ar  = wmma_f16(a1, loadB64(Whh0h, g0r, 1), ar);
            az  = wmma_f16(a0, loadB64(Whh0h, g0z, 0), az);
            az  = wmma_f16(a1, loadB64(Whh0h, g0z, 1), az);
            ahn = wmma_f16(a0, loadB64(Whh0h, g0n, 0), ahn);
            ahn = wmma_f16(a1, loadB64(Whh0h, g0n, 1), ahn);
        }
        __syncthreads();                 // everyone done reading h0 state
#pragma unroll
        for (int r = 0; r < 8; ++r) {
            int m = r + (lane >> 4) * 8;
            int col = wv * 16 + (lane & 15);
            float rv = sigm(ar[r]);
            float zv = sigm(az[r]);
            float nv = tanh_fast(axn[r] + rv * ahn[r]);
            float ho = h0s[m][col];
            float hn = (1.0f - zv) * nv + zv * ho;
            h0s[m][col] = hn;
            h0h[m][col] = (f16)hn;
        }
        __syncthreads();

        // ---------------- layer 1 ----------------
        ar = bcast8(b1r); az = bcast8(b1z); axn = bcast8(b1xn); ahn = bcast8(b1hn);
        {
            v16h x0 = buildA(h0h, 0), x1 = buildA(h0h, 1);     // layer0 output is the input
            ar  = wmma_f16(x0, loadB64(Wih1h, g0r, 0), ar);
            ar  = wmma_f16(x1, loadB64(Wih1h, g0r, 1), ar);
            az  = wmma_f16(x0, loadB64(Wih1h, g0z, 0), az);
            az  = wmma_f16(x1, loadB64(Wih1h, g0z, 1), az);
            axn = wmma_f16(x0, loadB64(Wih1h, g0n, 0), axn);
            axn = wmma_f16(x1, loadB64(Wih1h, g0n, 1), axn);
            v16h a0 = buildA(h1h, 0), a1 = buildA(h1h, 1);
            ar  = wmma_f16(a0, loadB64(Whh1h, g0r, 0), ar);
            ar  = wmma_f16(a1, loadB64(Whh1h, g0r, 1), ar);
            az  = wmma_f16(a0, loadB64(Whh1h, g0z, 0), az);
            az  = wmma_f16(a1, loadB64(Whh1h, g0z, 1), az);
            ahn = wmma_f16(a0, loadB64(Whh1h, g0n, 0), ahn);
            ahn = wmma_f16(a1, loadB64(Whh1h, g0n, 1), ahn);
        }
        __syncthreads();
#pragma unroll
        for (int r = 0; r < 8; ++r) {
            int m = r + (lane >> 4) * 8;
            int col = wv * 16 + (lane & 15);
            float rv = sigm(ar[r]);
            float zv = sigm(az[r]);
            float nv = tanh_fast(axn[r] + rv * ahn[r]);
            float ho = h1s[m][col];
            float hn = (1.0f - zv) * nv + zv * ho;
            h1s[m][col] = hn;
            h1h[m][col] = (f16)hn;
        }
        __syncthreads();
    }

    // ---------------- fused fc_in epilogue ----------------
    for (int o = tid; o < 16 * 64; o += 128) {
        int m = o >> 6, g = o & 63;
        float acc = fcb[g];
        const float* wr = fcW + g * 64;
#pragma unroll 16
        for (int h = 0; h < 64; ++h) acc += h1s[m][h] * wr[h];
        hidden_out[(size_t)(s0 + m) * 64 + g] = acc;
    }
}

// =====================================================================
// GAT scores: si/sj via folded vectors; also emit transposed f16 hidden
// (hT[h][s]) so WMMA B operands are contiguous 32B per lane.
// =====================================================================
__global__ __launch_bounds__(256)
void score_kernel(const float* __restrict__ hid, const float* __restrict__ va1,
                  const float* __restrict__ va2, const float* __restrict__ c12,
                  float* __restrict__ si, float* __restrict__ sj, f16* __restrict__ hT)
{
    int s = blockIdx.x * 256 + threadIdx.x;        // 0..8191
    const float* hp = hid + (size_t)s * 64;
    float a1 = c12[0], a2 = c12[1];
#pragma unroll 16
    for (int h = 0; h < 64; ++h) {
        float v = hp[h];
        a1 += v * va1[h];
        a2 += v * va2[h];
        hT[(size_t)h * 8192 + s] = (f16)v;
    }
    sj[s] = a1;
    si[s] = a2;
}

__global__ __launch_bounds__(256)
void maxsj_kernel(const float* __restrict__ sj, float* __restrict__ maxsj)
{
    __shared__ float red[256];
    int b = blockIdx.x, tid = threadIdx.x;
    float m = -3.4e38f;
    for (int j = tid; j < CN; j += 256) m = fmaxf(m, sj[b * CN + j]);
    red[tid] = m;
    __syncthreads();
    for (int off = 128; off > 0; off >>= 1) {
        if (tid < off) red[tid] = fmaxf(red[tid], red[tid + off]);
        __syncthreads();
    }
    if (tid == 0) maxsj[b] = red[0];
}

// =====================================================================
// Flash-style GAT aggregation: one block per (batch, 16-row tile).
// Row max is exactly LR(si + max_j sj) since leaky-relu is monotone.
// P tiles (f16) x hidden tiles (f16) via WMMA; denominator in VALU.
// =====================================================================
__global__ __launch_bounds__(128)
void gat_agg_kernel(const float* __restrict__ si, const float* __restrict__ sj,
                    const float* __restrict__ maxsj, const f16* __restrict__ hT,
                    const float* __restrict__ hin, float* __restrict__ hout)
{
    __shared__ float accL[4][16][68];
    __shared__ float dpart[4][32];
    __shared__ float dinv[16];

    const int tid = threadIdx.x, lane = tid & 31, wv = tid >> 5;
    const int b  = blockIdx.x >> 7;            // 128 row-tiles per batch
    const int i0 = (blockIdx.x & 127) * 16;

    const int mA  = lane & 15;
    const int kbA = (lane >> 4) * 16;
    const int nB  = lane & 15;
    const int kbB = (lane >> 4) * 16;

    const float mx   = maxsj[b];
    const float siv  = si[b * CN + i0 + mA];
    const float mrow = lrelu(siv + mx);

    const f16*  hTb = hT + b * CN;
    const float* sjb = sj + b * CN;

    v8f acc0 = bcast8(0.0f), acc1 = bcast8(0.0f), acc2 = bcast8(0.0f), acc3 = bcast8(0.0f);
    float dsum = 0.0f;

    for (int c = wv; c < CN / 32; c += 4) {
        int j0 = c * 32;
        v16h a;
#pragma unroll
        for (int i = 0; i < 16; ++i) {
            int j = j0 + kbA + i;
            float p = __expf(lrelu(siv + sjb[j]) - mrow);
            dsum += p;
            a[i] = (f16)p;
        }
        acc0 = wmma_f16(a, *(const v16h*)(hTb + (size_t)(0  + nB) * 8192 + j0 + kbB), acc0);
        acc1 = wmma_f16(a, *(const v16h*)(hTb + (size_t)(16 + nB) * 8192 + j0 + kbB), acc1);
        acc2 = wmma_f16(a, *(const v16h*)(hTb + (size_t)(32 + nB) * 8192 + j0 + kbB), acc2);
        acc3 = wmma_f16(a, *(const v16h*)(hTb + (size_t)(48 + nB) * 8192 + j0 + kbB), acc3);
    }

#pragma unroll
    for (int r = 0; r < 8; ++r) {
        int m = r + (lane >> 4) * 8, n = lane & 15;
        accL[wv][m][n]      = acc0[r];
        accL[wv][m][16 + n] = acc1[r];
        accL[wv][m][32 + n] = acc2[r];
        accL[wv][m][48 + n] = acc3[r];
    }
    dpart[wv][lane] = dsum;
    __syncthreads();
    if (tid < 16) {
        float d = 0.0f;
#pragma unroll
        for (int w = 0; w < 4; ++w) d += dpart[w][tid] + dpart[w][tid + 16];
        dinv[tid] = __builtin_amdgcn_rcpf(d);     // denom >= 1 always (max term present)
    }
    __syncthreads();
    for (int o = tid; o < 16 * 64; o += 128) {
        int m = o >> 6, h = o & 63;
        float v = accL[0][m][h] + accL[1][m][h] + accL[2][m][h] + accL[3][m][h];
        size_t gi = (size_t)(b * CN + i0 + m) * 64 + h;
        hout[gi] = v * dinv[m] + hin[gi];
    }
}

// =====================================================================
// FFN head: out = LR(h @ Wffn^T + b) @ Wout^T + bout
// =====================================================================
__global__ __launch_bounds__(128)
void ffn_kernel(const float* __restrict__ hid, const float* __restrict__ ffnW,
                const float* __restrict__ ffnb, const float* __restrict__ foW,
                const float* __restrict__ fob, float* __restrict__ out)
{
    int s = blockIdx.x * 128 + threadIdx.x;
    const float* hp = hid + (size_t)s * 64;
    float row[64];
#pragma unroll
    for (int h = 0; h < 64; ++h) row[h] = hp[h];
    float o = fob[0];
    for (int g = 0; g < 64; ++g) {
        float acc = ffnb[g];
        const float* wr = ffnW + g * 64;
#pragma unroll 16
        for (int h = 0; h < 64; ++h) acc += row[h] * wr[h];
        o += foW[g] * lrelu(acc);
    }
    out[s] = o;
}

// =====================================================================
extern "C" void kernel_launch(void* const* d_in, const int* in_sizes, int n_in,
                              void* d_out, int out_size, void* d_ws, size_t ws_size,
                              hipStream_t stream)
{
    const float* x      = (const float*)d_in[0];
    const float* Wih0   = (const float*)d_in[1];
    const float* Whh0   = (const float*)d_in[2];
    const float* bih0   = (const float*)d_in[3];
    const float* bhh0   = (const float*)d_in[4];
    const float* Wih1   = (const float*)d_in[5];
    const float* Whh1   = (const float*)d_in[6];
    const float* bih1   = (const float*)d_in[7];
    const float* bhh1   = (const float*)d_in[8];
    const float* fcW    = (const float*)d_in[9];
    const float* fcb    = (const float*)d_in[10];
    const float* transW = (const float*)d_in[11];
    const float* transb = (const float*)d_in[12];
    const float* av     = (const float*)d_in[13];
    const float* ffnW   = (const float*)d_in[14];
    const float* ffnb   = (const float*)d_in[15];
    const float* foW    = (const float*)d_in[16];
    const float* fob    = (const float*)d_in[17];

    char* ws = (char*)d_ws;
    size_t off = 0;
    auto walloc = [&](size_t bytes) -> void* {
        void* p = ws + off;
        off = (off + bytes + 255) & ~(size_t)255;
        return p;
    };
    float* hidA  = (float*)walloc((size_t)8192 * 64 * 4);
    float* hidB  = (float*)walloc((size_t)8192 * 64 * 4);
    float* si    = (float*)walloc(8192 * 4);
    float* sj    = (float*)walloc(8192 * 4);
    float* mxsj  = (float*)walloc(4 * 4);
    float* va1   = (float*)walloc(64 * 4);
    float* va2   = (float*)walloc(64 * 4);
    float* c12   = (float*)walloc(2 * 4);
    f16*  hT     = (f16*)walloc((size_t)64 * 8192 * 2);
    f16*  Wih0h  = (f16*)walloc(192 * 32 * 2);
    f16*  Whh0h  = (f16*)walloc(192 * 64 * 2);
    f16*  Wih1h  = (f16*)walloc(192 * 64 * 2);
    f16*  Whh1h  = (f16*)walloc(192 * 64 * 2);
    (void)in_sizes; (void)n_in; (void)out_size; (void)ws_size;

    prep_kernel<<<1, 256, 0, stream>>>(Wih0, Whh0, Wih1, Whh1, transW, transb, av,
                                       Wih0h, Whh0h, Wih1h, Whh1h, va1, va2, c12);

    gru_fused_kernel<<<512, 128, 0, stream>>>(x, Wih0h, Whh0h, Wih1h, Whh1h,
                                              bih0, bhh0, bih1, bhh1, fcW, fcb, hidA);

    float* cur = hidA;
    float* nxt = hidB;
    for (int l = 0; l < 2; ++l) {
        score_kernel<<<32, 256, 0, stream>>>(cur, va1, va2, c12, si, sj, hT);
        maxsj_kernel<<<4, 256, 0, stream>>>(sj, mxsj);
        gat_agg_kernel<<<512, 128, 0, stream>>>(si, sj, mxsj, hT, cur, nxt);
        float* tmp = cur; cur = nxt; nxt = tmp;
    }

    ffn_kernel<<<64, 128, 0, stream>>>(cur, ffnW, ffnb, foW, fob, (float*)d_out);
}
